// Attention_80968723464310
// MI455X (gfx1250) — compile-verified, matrix operand order checked by reading, not run on hardware
//
#include <hip/hip_runtime.h>
#include <math.h>

#define HID   512
#define NB    16
#define LSEQ  2048
#define TWOH  1024
#define MT    128
#define LDA   1032   // bf16 row stride: 2064 B = 129*16 -> 16B-aligned rows

typedef __bf16 bf16_t;
typedef bf16_t v8bf  __attribute__((ext_vector_type(8)));
typedef bf16_t v16bf __attribute__((ext_vector_type(16)));
typedef float  v8f   __attribute__((ext_vector_type(8)));

__device__ __forceinline__ unsigned short f32_to_bf16(float f) {
  union { float f; unsigned u; } x; x.f = f;
  unsigned r = x.u + 0x7FFFu + ((x.u >> 16) & 1u);   // round-to-nearest-even
  return (unsigned short)(r >> 16);
}

// CDNA5 hardware tanh (TRANS32 op) -- branch-free, replaces libm's exp-based
// expansion that introduced EXEC divergence between WMMA loop trips.
__device__ __forceinline__ float htanh(float x) {
  float y;
  asm("v_tanh_f32 %0, %1" : "=v"(y) : "v"(x));
  return y;
}

// ---------------- kernel 0: qb1[b,j] = b1[j] + sum_i q[b,i]*W1q[i,j] ----------------
__global__ void qproj_kernel(const float* __restrict__ query,
                             const float* __restrict__ W1,
                             const float* __restrict__ b1,
                             float* __restrict__ qb1) {
  __shared__ float qs[TWOH];
  const int b = blockIdx.x, tid = threadIdx.x;
  for (int i = tid; i < TWOH; i += 256) qs[i] = query[b * TWOH + i];
  __syncthreads();
  float acc[4];
#pragma unroll
  for (int r = 0; r < 4; ++r) acc[r] = b1[tid + r * 256];
  for (int i = 0; i < TWOH; ++i) {
    const float q = qs[i];
    const float* row = W1 + (size_t)i * TWOH;   // W1q rows 0..1023
#pragma unroll
    for (int r = 0; r < 4; ++r) acc[r] = fmaf(q, row[tid + r * 256], acc[r]);
  }
#pragma unroll
  for (int r = 0; r < 4; ++r) qb1[b * TWOH + tid + r * 256] = acc[r];
}

// ---------------- kernel 1: fused keys@W1k GEMM (bf16 WMMA) + tanh + dot(W2) ----------------
// grid: (LSEQ/MT, NB), block 256 (8 waves), dynamic LDS ~298 KB.
__global__ void scores_kernel(const float* __restrict__ keys,
                              const float* __restrict__ coverage,
                              const float* __restrict__ W1,
                              const float* __restrict__ W2,
                              const float* __restrict__ qb1,
                              float* __restrict__ scores) {
  extern __shared__ char smem[];
  unsigned short* As  = (unsigned short*)smem;                          // [MT][LDA] bf16
  unsigned short* Bs  = (unsigned short*)(smem + (size_t)MT * LDA * 2); // [16][LDA] bf16
  float* covs = (float*)(smem + (size_t)(MT + 16) * LDA * 2);           // [MT]
  float* ssum = covs + MT;                                              // [MT]

  const int b   = blockIdx.y;
  const int m0  = blockIdx.x * MT;
  const int tid = threadIdx.x;
  const int w    = tid >> 5;     // wave 0..7 -> rows w*16..w*16+15
  const int lane = tid & 31;
  const int hl   = lane >> 4;    // lane half selects K group (+8)
  const int ln   = lane & 15;    // A: row M; B: col N; C/D: col N

  // stage A tile (keys rows, fp32 -> bf16)
  const float* kbase = keys + ((size_t)b * LSEQ + m0) * TWOH;
  for (int e = tid; e < MT * TWOH; e += 256) {
    const int row = e >> 10, col = e & (TWOH - 1);
    As[row * LDA + col] = f32_to_bf16(kbase[(size_t)row * TWOH + col]);
  }
  if (tid < MT) {
    covs[tid] = coverage[(size_t)b * LSEQ + m0 + tid];
    ssum[tid] = 0.0f;
  }

  const float* W1k = W1 + (size_t)TWOH * TWOH;   // rows 1024..2047
  const float* W1c = W1 + (size_t)2048 * TWOH;   // row  2048

  float sacc[8];
#pragma unroll
  for (int r = 0; r < 8; ++r) sacc[r] = 0.0f;

  const unsigned short* arow = As + (w * 16 + ln) * LDA + hl * 8;
  const unsigned short* brow = Bs + ln * LDA + hl * 8;

  for (int nc = 0; nc < TWOH / 16; ++nc) {
    const int n0 = nc * 16;
    __syncthreads();  // previous chunk's compute done before B overwrite
    // stage B chunk: W1k[:,n0:n0+16] transposed -> Bs[n][k], bf16
    for (int e = tid; e < 16 * TWOH; e += 256) {
      const int n = e & 15, k = e >> 4;
      Bs[n * LDA + k] = f32_to_bf16(W1k[(size_t)k * TWOH + n0 + n]);
    }
    // overlap: prefetch next chunk's W1k columns into cache during compute
    if (nc + 1 < TWOH / 16) {
      const float* nxt = W1k + (size_t)0 * TWOH + (n0 + 16);
#pragma unroll
      for (int p = 0; p < 4; ++p)
        __builtin_prefetch(nxt + (size_t)(tid + p * 256) * TWOH, 0, 3);
    }
    __syncthreads();

    v8f c = {};
#pragma unroll 4
    for (int k0 = 0; k0 < TWOH; k0 += 32) {
      const v8bf alo = *(const v8bf*)(arow + k0);        // K = k0 + hl*8 .. +7
      const v8bf ahi = *(const v8bf*)(arow + k0 + 16);   // K = k0+16+hl*8 .. +7
      const v8bf blo = *(const v8bf*)(brow + k0);
      const v8bf bhi = *(const v8bf*)(brow + k0 + 16);
      const v16bf a  = __builtin_shufflevector(alo, ahi, 0,1,2,3,4,5,6,7,8,9,10,11,12,13,14,15);
      const v16bf bb = __builtin_shufflevector(blo, bhi, 0,1,2,3,4,5,6,7,8,9,10,11,12,13,14,15);
      c = __builtin_amdgcn_wmma_f32_16x16x32_bf16(false, a, false, bb, (short)0, c, false, false);
    }

    // fused epilogue: v_tanh_f32(C + qb1[n] + cov*W1c[n]) * W2[n]
    const int n = n0 + ln;
    const float qv  = qb1[b * TWOH + n];
    const float w1c = W1c[n];
    const float w2n = W2[n];
#pragma unroll
    for (int r = 0; r < 8; ++r) {
      const int mrow = w * 16 + r + hl * 8;   // C/D layout: lanes 16-31 hold M+8
      const float pre = c[r] + qv + covs[mrow] * w1c;
      sacc[r] += htanh(pre) * w2n;
    }
  }

#pragma unroll
  for (int r = 0; r < 8; ++r)
    atomicAdd(&ssum[w * 16 + r + hl * 8], sacc[r]);  // 16-lane n-slice reduction
  __syncthreads();
  if (tid < MT) scores[(size_t)b * LSEQ + m0 + tid] = ssum[tid];
}

// ---------------- kernel 2: softmax over L per batch ----------------
__global__ void softmax_kernel(const float* __restrict__ scores,
                               float* __restrict__ att) {
  __shared__ float red[256];
  __shared__ float stats[2];
  const int b = blockIdx.x, tid = threadIdx.x;
  const float* s = scores + (size_t)b * LSEQ;
  float v[8];
  float m = -INFINITY;
#pragma unroll
  for (int i = 0; i < 8; ++i) { v[i] = s[tid + i * 256]; m = fmaxf(m, v[i]); }
  red[tid] = m; __syncthreads();
  for (int off = 128; off > 0; off >>= 1) {
    if (tid < off) red[tid] = fmaxf(red[tid], red[tid + off]);
    __syncthreads();
  }
  if (tid == 0) stats[0] = red[0];
  __syncthreads();
  m = stats[0];
  float sum = 0.0f;
#pragma unroll
  for (int i = 0; i < 8; ++i) { v[i] = __expf(v[i] - m); sum += v[i]; }
  red[tid] = sum; __syncthreads();
  for (int off = 128; off > 0; off >>= 1) {
    if (tid < off) red[tid] += red[tid + off];
    __syncthreads();
  }
  if (tid == 0) stats[1] = red[0];
  __syncthreads();
  const float inv = 1.0f / stats[1];
#pragma unroll
  for (int i = 0; i < 8; ++i) att[(size_t)b * LSEQ + tid + i * 256] = v[i] * inv;
}

// ---------------- kernel 3: t[b,:] = att[b,:]^T @ keys[b]  (L2-resident 2nd pass) ----------------
__global__ void wsum_kernel(const float* __restrict__ keys,
                            const float* __restrict__ att,
                            float* __restrict__ t) {
  const int b = blockIdx.y;
  const int c = blockIdx.x * 256 + threadIdx.x;
  const float* kb = keys + (size_t)b * LSEQ * TWOH + c;
  const float* ab = att + (size_t)b * LSEQ;
  float acc = 0.0f;
  for (int l = 0; l < LSEQ; ++l) acc = fmaf(ab[l], kb[(size_t)l * TWOH], acc);
  t[b * TWOH + c] = acc;
}

// ---------------- kernel 4: context[b,:] = t[b,:] @ Wr ----------------
__global__ void ctx_kernel(const float* __restrict__ t,
                           const float* __restrict__ Wr,
                           float* __restrict__ ctx) {
  __shared__ float ts[TWOH];
  const int b = blockIdx.y;
  const int o = blockIdx.x * 256 + threadIdx.x;
  for (int i = threadIdx.x; i < TWOH; i += 256) ts[i] = t[b * TWOH + i];
  __syncthreads();
  float acc = 0.0f;
  for (int j = 0; j < TWOH; ++j) acc = fmaf(ts[j], Wr[(size_t)j * HID + o], acc);
  ctx[b * HID + o] = acc;
}

extern "C" void kernel_launch(void* const* d_in, const int* in_sizes, int n_in,
                              void* d_out, int out_size, void* d_ws, size_t ws_size,
                              hipStream_t stream) {
  const float* query    = (const float*)d_in[0];  // (16,1,1024)
  const float* keys     = (const float*)d_in[1];  // (16,2048,1024)
  const float* coverage = (const float*)d_in[2];  // (16,2048,1)
  const float* W1       = (const float*)d_in[3];  // (2049,1024)
  const float* b1       = (const float*)d_in[4];  // (1024,)
  const float* W2       = (const float*)d_in[5];  // (1024,1)
  const float* Wr       = (const float*)d_in[6];  // (1024,512)

  float* ctx = (float*)d_out;             // context: 16*512 floats
  float* att = (float*)d_out + NB * HID;  // attention: 16*2048 floats

  float* qb1    = (float*)d_ws;           // 16*1024
  float* scores = qb1 + NB * TWOH;        // 16*2048
  float* t      = scores + NB * LSEQ;     // 16*1024

  const size_t smem = (size_t)(MT + 16) * LDA * 2 + (size_t)MT * 8;  // 298,240 B
  (void)hipFuncSetAttribute((const void*)scores_kernel,
                            hipFuncAttributeMaxDynamicSharedMemorySize, (int)smem);

  qproj_kernel<<<NB, 256, 0, stream>>>(query, W1, b1, qb1);
  scores_kernel<<<dim3(LSEQ / MT, NB), 256, smem, stream>>>(keys, coverage, W1, W2, qb1, scores);
  softmax_kernel<<<NB, 256, 0, stream>>>(scores, att);
  wsum_kernel<<<dim3(TWOH / 256, NB), 256, 0, stream>>>(keys, att, t);
  ctx_kernel<<<dim3(HID / 256, NB), 256, 0, stream>>>(t, Wr, ctx);
}
